// STSGCN_Transformer_22333829939229
// MI455X (gfx1250) — matrix operand order repaired and checked by reading, not compiled, not run on hardware
//
#include <hip/hip_runtime.h>

// ---------------------------------------------------------------------------
// STSGCN + Transformer decoder for MI455X (gfx1250, wave32, WMMA)
// B=2048, T_IN=10, T_OUT=25, V=22, C=3, D=66, H=3, HD=22
// All GEMMs: zero-padded K (multiple of 32), M multiple of 16, packed f16
// weights in WMMA B-fragment order -> branch-free WMMA inner loops.
// GCN conv folds all N-tiles into one wave (A fragments fetched once).
// ---------------------------------------------------------------------------

#define BATCH   2048
#define T_IN    10
#define T_OUT   25
#define NV      22
#define NC      3
#define ND      66          // V*C
#define NH      3
#define HD      22
#define TV      220         // T_IN * NV
#define MROWS   (BATCH*TV)  // 450560, multiple of 16
#define EPSV    1e-5f

typedef __attribute__((ext_vector_type(16))) _Float16 v16h;
typedef __attribute__((ext_vector_type(8)))  float    v8f;
typedef __attribute__((ext_vector_type(4)))  float    v4f;

// ---------------------------------------------------------------------------
// Pack fp32 weight W (N x K row-major) into f16 WMMA B-fragment order:
// idx = ((nt*KK + kk)*32 + lane)*16 + j ; col = nt*16+(lane&15),
// k = kk*32 + (j<8 ? (lane>>4)*8 + j : 16 + (lane>>4)*8 + j-8) ; zero outside (N,K).
// ---------------------------------------------------------------------------
__global__ void pack_w(const float* __restrict__ W, _Float16* __restrict__ out,
                       int N, int K, int KK, int total)
{
    int idx = blockIdx.x * blockDim.x + threadIdx.x;
    if (idx >= total) return;
    int j    = idx & 15;
    int lane = (idx >> 4) & 31;
    int blk  = idx >> 9;              // nt*KK + kk
    int kk   = blk % KK;
    int nt   = blk / KK;
    int col  = nt * 16 + (lane & 15);
    int k    = kk * 32 + (j < 8 ? (lane >> 4) * 8 + j : 16 + (lane >> 4) * 8 + (j - 8));
    float v  = (col < N && k < K) ? W[(size_t)col * K + k] : 0.f;
    out[idx] = (_Float16)v;
}

// Load an A fragment (16x32 f16 WMMA layout) from a padded fp32 row pointer.
__device__ __forceinline__ v16h load_a_frag(const float* __restrict__ ap)
{
    v4f p0 = *(const v4f*)(ap);
    v4f p1 = *(const v4f*)(ap + 4);
    v4f p2 = *(const v4f*)(ap + 16);
    v4f p3 = *(const v4f*)(ap + 20);
    v16h a;
#pragma unroll
    for (int j = 0; j < 4; ++j) {
        a[j]      = (_Float16)p0[j];
        a[4 + j]  = (_Float16)p1[j];
        a[8 + j]  = (_Float16)p2[j];
        a[12 + j] = (_Float16)p3[j];
    }
    return a;
}

// ---------------------------------------------------------------------------
// Branch-free WMMA GEMM: out[m,n] = (sum_k A[m,k]*W[n,k] + bias[n])*scale (+add)
// A: (Mtiles*16) x lda fp32, lda multiple of 32, K-pad columns are zero.
// Wpk: packed f16 fragments. outmode: 0 = skip n>=N, 1 = store 0 for n>=N.
// out2 (optional) always stores 0 for n>=N (keeps padded state buffers clean).
// ---------------------------------------------------------------------------
__global__ void gemm_wmma(const float* __restrict__ A, int lda,
                          const _Float16* __restrict__ Wpk,
                          const float* __restrict__ bias,
                          const float* __restrict__ add, int ldadd,
                          float* __restrict__ out, int ldo, int outmode,
                          float* __restrict__ out2, int ldo2,
                          int Ntiles, int KK, int N, float scale, int totTiles)
{
    const int wave = threadIdx.x >> 5;
    const int lane = threadIdx.x & 31;
    const int tile = blockIdx.x * (blockDim.x >> 5) + wave;
    if (tile >= totTiles) return;
    const int mt = tile / Ntiles, nt = tile % Ntiles;
    const int hl = lane >> 4;
    const int ll = lane & 15;

    const float*    ap = A + (size_t)(mt * 16 + ll) * lda + hl * 8;
    const _Float16* wp = Wpk + ((size_t)nt * KK * 32 + lane) * 16;

    v8f acc = {};
    for (int kk = 0; kk < KK; ++kk) {
        v16h a = load_a_frag(ap);
        ap += 32;
        v16h b = *(const v16h*)(wp);
        wp += 512;
        acc = __builtin_amdgcn_wmma_f32_16x16x32_f16(false, a, false, b,
                                                     (short)0, acc, false, false);
    }

    const int bcol = nt * 16 + ll;
    const bool inN = bcol < N;
    const float bv = (bias && inN) ? bias[bcol] : 0.f;
#pragma unroll
    for (int r = 0; r < 8; ++r) {
        int m = mt * 16 + r + hl * 8;        // C layout: M = r + 8*(lane/16)
        float y = (acc[r] + bv) * scale;
        if (add) y += add[(size_t)m * ldadd + bcol];
        float yz = inN ? y : 0.f;
        if (outmode == 1)      out[(size_t)m * ldo + bcol] = yz;
        else if (inN)          out[(size_t)m * ldo + bcol] = y;
        if (out2)              out2[(size_t)m * ldo2 + bcol] = yz;
    }
}

// ---------------------------------------------------------------------------
// Fused GCN conv (channel-last): rows r = b*220+tv, A-layout (MROWS, NT*16? no:
// cipad = KK*32).  One wave owns ALL NT output-channel tiles of one 16-row
// M-tile: z/h fragments are loaded once per k-step and reused 2*NT times.
// out = leaky( BN(z@cw.T+cb) + BN(h@rw.T+rb), pre ), out (MROWS, NT*16),
// pad output channels written as zero.
// ---------------------------------------------------------------------------
template <int NT, int KK>
__global__ void gcn_conv_wmma(const float* __restrict__ z, const float* __restrict__ hres,
                              const _Float16* __restrict__ cwp, const _Float16* __restrict__ rwp,
                              const float* __restrict__ cb, const float* __restrict__ g,
                              const float* __restrict__ bt, const float* __restrict__ rb,
                              const float* __restrict__ rg, const float* __restrict__ rbt,
                              const float* __restrict__ pre, float* __restrict__ out,
                              int co)
{
    const int wave = threadIdx.x >> 5;
    const int lane = threadIdx.x & 31;
    const int mt = blockIdx.x * (blockDim.x >> 5) + wave;
    if (mt >= MROWS / 16) return;
    const int hl = lane >> 4;
    const int ll = lane & 15;
    const int cipad = KK * 32;
    const int copad = NT * 16;

    const float* ap1 = z    + (size_t)(mt * 16 + ll) * cipad + hl * 8;
    const float* ap2 = hres + (size_t)(mt * 16 + ll) * cipad + hl * 8;
    const _Float16* wp1 = cwp + (size_t)lane * 16;
    const _Float16* wp2 = rwp + (size_t)lane * 16;

    v8f am[NT], ar[NT];
#pragma unroll
    for (int n = 0; n < NT; ++n) { am[n] = (v8f){}; ar[n] = (v8f){}; }

#pragma unroll
    for (int kk = 0; kk < KK; ++kk) {
        v16h a1 = load_a_frag(ap1);
        v16h a2 = load_a_frag(ap2);
        ap1 += 32; ap2 += 32;
#pragma unroll
        for (int n = 0; n < NT; ++n) {
            v16h b1 = *(const v16h*)(wp1 + ((size_t)n * KK + kk) * 512);
            v16h b2 = *(const v16h*)(wp2 + ((size_t)n * KK + kk) * 512);
            am[n] = __builtin_amdgcn_wmma_f32_16x16x32_f16(false, a1, false, b1,
                                                           (short)0, am[n], false, false);
            ar[n] = __builtin_amdgcn_wmma_f32_16x16x32_f16(false, a2, false, b2,
                                                           (short)0, ar[n], false, false);
        }
    }

    const float inv = rsqrtf(1.f + EPSV);
    const float prew = pre[0];
#pragma unroll
    for (int n = 0; n < NT; ++n) {
        const int ob = n * 16 + ll;
        const bool inN = ob < co;
        float cbv = 0.f, gv = 0.f, btv = 0.f, rbv = 0.f, rgv = 0.f, rbtv = 0.f;
        if (inN) {
            cbv = cb[ob]; gv = g[ob] * inv; btv = bt[ob];
            rbv = rb[ob]; rgv = rg[ob] * inv; rbtv = rbt[ob];
        }
#pragma unroll
        for (int r = 0; r < 8; ++r) {
            int m = mt * 16 + r + hl * 8;
            float ym = (am[n][r] + cbv) * gv + btv;
            float yr = (ar[n][r] + rbv) * rgv + rbtv;
            float y = ym + yr;
            y = (y >= 0.f) ? y : prew * y;
            out[(size_t)m * copad + ob] = inN ? y : 0.f;
        }
    }
}

// ---------------------------------------------------------------------------
// x (B,T,V,C) -> h0 channel-last (B*220, 32), pad channels zero.
// ---------------------------------------------------------------------------
__global__ void transpose_in(const float* __restrict__ x, float* __restrict__ h0)
{
    int idx = blockIdx.x * blockDim.x + threadIdx.x;
    if (idx >= MROWS * 32) return;
    int r = idx >> 5, c = idx & 31;      // r = b*220 + t*22 + v
    int b = r / TV, tv = r % TV;
    int t = tv / NV, v = tv % NV;
    float val = (c < NC) ? x[(((size_t)b * T_IN + t) * NV + v) * NC + c] : 0.f;
    h0[idx] = val;
}

// ---------------------------------------------------------------------------
// Graph mixes (one block per sample, 8-channel groups through LDS):
// y2[tv', c] = sum_v (sum_t h[t,v,c]*Tm[v,t,q]) * A[q,v,w]   (channel-last)
// Pad channels of h are zero => pad outputs are zero automatically.
// ---------------------------------------------------------------------------
__global__ void gcn_spatial(const float* __restrict__ h, float* __restrict__ y2,
                            const float* __restrict__ Amat, const float* __restrict__ Tm,
                            int cipad)
{
    __shared__ float hb[8 * TV];
    __shared__ float y1[8 * TV];
    const int b = blockIdx.x;
    const int tid = threadIdx.x;
    const float* src = h  + (size_t)b * TV * cipad;
    float*       dst = y2 + (size_t)b * TV * cipad;

    for (int c0 = 0; c0 < cipad; c0 += 8) {
        for (int idx = tid; idx < 8 * TV; idx += 256) {
            int c = idx & 7, tv = idx >> 3;
            hb[c * TV + tv] = src[(size_t)tv * cipad + c0 + c];
        }
        __syncthreads();
        for (int idx = tid; idx < 8 * TV; idx += 256) {
            int c = idx / TV, r = idx % TV;
            int q = r / NV, v = r % NV;
            float s = 0.f;
#pragma unroll
            for (int tt = 0; tt < T_IN; ++tt)
                s += hb[c * TV + tt * NV + v] * Tm[v * T_IN * T_IN + tt * T_IN + q];
            y1[idx] = s;
        }
        __syncthreads();
        for (int idx = tid; idx < 8 * TV; idx += 256) {
            int c = idx & 7, r = idx >> 3;   // r = t*22 + w
            int t = r / NV, w = r % NV;
            float s = 0.f;
#pragma unroll
            for (int v = 0; v < NV; ++v)
                s += y1[c * TV + t * NV + v] * Amat[t * NV * NV + v * NV + w];
            dst[(size_t)r * cipad + c0 + c] = s;
        }
        __syncthreads();
    }
}

// enc[b*10+t][d<96] ; d<66: hfin[(b*220+t*22+v)*32 + c], else 0
__global__ void make_enc(const float* __restrict__ hfin, float* __restrict__ enc)
{
    int idx = blockIdx.x * blockDim.x + threadIdx.x;
    if (idx >= BATCH * T_IN * 96) return;
    int row = idx / 96, d = idx % 96;
    float val = 0.f;
    if (d < ND) {
        int b = row / T_IN, t = row % T_IN;
        int v = d / NC, c = d % NC;
        val = hfin[((size_t)b * TV + t * NV + v) * 32 + c];
    }
    enc[idx] = val;
}

// hs = enc[:, -1, :] (padded-zero), tp = x[:, -1, :] (padded-zero)
__global__ void init_state(const float* __restrict__ enc, const float* __restrict__ x,
                           float* __restrict__ hs, float* __restrict__ tp)
{
    int idx = blockIdx.x * blockDim.x + threadIdx.x;
    if (idx >= BATCH * 96) return;
    int b = idx / 96, d = idx % 96;
    hs[idx] = enc[((size_t)b * T_IN + (T_IN - 1)) * 96 + d];   // pads already zero
    tp[idx] = (d < ND) ? x[(size_t)b * T_IN * ND + (T_IN - 1) * ND + d] : 0.f;
}

// GRU gates: hs(96-pad) = (1-z)*n + z*hs ; cat[:,0:66]=hs ; cat[:,146:160]=0
__global__ void gru_gates(const float* __restrict__ gi, const float* __restrict__ gh,
                          float* __restrict__ hs, float* __restrict__ cat)
{
    int idx = blockIdx.x * blockDim.x + threadIdx.x;
    if (idx >= BATCH * ND) return;
    int b = idx / ND, d = idx % ND;
    const float* gib = gi + (size_t)b * 3 * ND;
    const float* ghb = gh + (size_t)b * 3 * ND;
    float r = 1.f / (1.f + __expf(-(gib[d] + ghb[d])));
    float z = 1.f / (1.f + __expf(-(gib[ND + d] + ghb[ND + d])));
    float n = tanhf(gib[2 * ND + d] + r * ghb[2 * ND + d]);
    float h = hs[(size_t)b * 96 + d];
    float hn = (1.f - z) * n + z * h;
    hs[(size_t)b * 96 + d] = hn;
    cat[(size_t)b * 160 + d] = hn;
    if (d < 14) cat[(size_t)b * 160 + 146 + d] = 0.f;  // tail pad of K=132->160
}

// Single-query attention; ctx padded to 96 cols (pads zeroed by head 0).
__global__ void attn_kernel(const float* __restrict__ q, const float* __restrict__ k,
                            const float* __restrict__ v, float* __restrict__ ctx)
{
    int idx = blockIdx.x * blockDim.x + threadIdx.x;
    if (idx >= BATCH * NH) return;
    int b = idx / NH, h = idx % NH;
    const float* qp = q + (size_t)b * ND + h * HD;
    float sc[T_IN];
    float mx = -1e30f;
#pragma unroll
    for (int t = 0; t < T_IN; ++t) {
        const float* kp = k + ((size_t)b * T_IN + t) * ND + h * HD;
        float s = 0.f;
#pragma unroll
        for (int d = 0; d < HD; ++d) s += qp[d] * kp[d];
        sc[t] = s;
        mx = fmaxf(mx, s);
    }
    float sum = 0.f;
#pragma unroll
    for (int t = 0; t < T_IN; ++t) { float e = __expf(sc[t] - mx); sc[t] = e; sum += e; }
    float inv = 1.f / sum;
#pragma unroll
    for (int d = 0; d < HD; ++d) {
        float c = 0.f;
#pragma unroll
        for (int t = 0; t < T_IN; ++t)
            c += sc[t] * v[((size_t)b * T_IN + t) * ND + h * HD + d];
        ctx[(size_t)b * 96 + h * HD + d] = c * inv;
    }
    if (h == 0)
        for (int p = ND; p < 96; ++p) ctx[(size_t)b * 96 + p] = 0.f;
}

// ---------------------------------------------------------------------------
// Host-side orchestration
// ---------------------------------------------------------------------------
extern "C" void kernel_launch(void* const* d_in, const int* in_sizes, int n_in,
                              void* d_out, int out_size, void* d_ws, size_t ws_size,
                              hipStream_t stream)
{
    (void)in_sizes; (void)n_in; (void)out_size; (void)ws_size;

    const float* x      = (const float*)d_in[0];
    const float* Aall   = (const float*)d_in[1];   // (4,10,22,22)
    const float* Tmall  = (const float*)d_in[2];   // (4,22,10,10)
    const float* pre    = (const float*)d_in[3];   // (4,)
    const float* gru_wi = (const float*)d_in[36];
    const float* gru_wh = (const float*)d_in[37];
    const float* gru_bi = (const float*)d_in[38];
    const float* gru_bh = (const float*)d_in[39];
    const float* qkv_w  = (const float*)d_in[40];  // (3D, D)
    const float* qkv_b  = (const float*)d_in[41];
    const float* ow     = (const float*)d_in[42];  // (D, D)
    const float* obias  = (const float*)d_in[43];
    const float* lw     = (const float*)d_in[44];  // (D, 2D)
    const float* lb     = (const float*)d_in[45];

    float* out = (float*)d_out;                    // (B, 25, 66)
    float* ws  = (float*)d_ws;

    // ---------------- workspace layout (floats) ----------------
    const size_t S = (size_t)MROWS * 64;           // 28,835,840 per encoder buffer
    float* buf0 = ws;
    float* buf1 = ws + S;
    float* buf2 = ws + 2 * S;
    float* enc  = ws + 3 * S;                               // (B*10, 96)
    float* kb   = enc + (size_t)BATCH * T_IN * 96;          // (B*10, 66)
    float* vb   = kb  + (size_t)BATCH * T_IN * ND;
    float* hs   = vb  + (size_t)BATCH * T_IN * ND;          // (B, 96)
    float* tp   = hs  + (size_t)BATCH * 96;
    float* gi   = tp  + (size_t)BATCH * 96;                 // (B, 198)
    float* gh   = gi  + (size_t)BATCH * 3 * ND;
    float* qb   = gh  + (size_t)BATCH * 3 * ND;             // (B, 66)
    float* ctx  = qb  + (size_t)BATCH * ND;                 // (B, 96)
    float* cat  = ctx + (size_t)BATCH * 96;                 // (B, 160)
    _Float16* pk = (_Float16*)(cat + (size_t)BATCH * 160);  // packed weights

    // packed weight slots (sizes in halves, all multiples of 512)
    _Float16* pk_wq = pk;             // 5*3*512
    _Float16* pk_wk = pk_wq + 7680;
    _Float16* pk_wv = pk_wk + 7680;
    _Float16* pk_ow = pk_wv + 7680;   // 5*3*512
    _Float16* pk_wi = pk_ow + 7680;   // 13*3*512
    _Float16* pk_wh = pk_wi + 19968;
    _Float16* pk_lw = pk_wh + 19968;  // 5*5*512
    _Float16* pk_cw[4], *pk_rw[4];
    {
        _Float16* p = pk_lw + 12800;
        for (int i = 0; i < 4; ++i) { pk_cw[i] = p; p += 2048; pk_rw[i] = p; p += 2048; }
    }

    auto pack = [&](const float* W, _Float16* dstp, int N, int K, int Ntiles, int KK) {
        int total = Ntiles * KK * 512;
        pack_w<<<(total + 255) / 256, 256, 0, stream>>>(W, dstp, N, K, KK, total);
    };
    auto gemm = [&](const float* A, int lda, const _Float16* Wp, const float* bias,
                    const float* add, int ldadd, float* o, int ldo, int omode,
                    float* o2, int ldo2, int M, int Ntiles, int KK, int N, float scale) {
        int tot = (M / 16) * Ntiles;
        gemm_wmma<<<(tot + 7) / 8, 256, 0, stream>>>(
            A, lda, Wp, bias, add, ldadd, o, ldo, omode, o2, ldo2,
            Ntiles, KK, N, scale, tot);
    };

    // ---------------- pack all weights ----------------
    pack(qkv_w,               pk_wq, ND, ND, 5, 3);
    pack(qkv_w + ND * ND,     pk_wk, ND, ND, 5, 3);
    pack(qkv_w + 2 * ND * ND, pk_wv, ND, ND, 5, 3);
    pack(ow,                  pk_ow, ND, ND, 5, 3);
    pack(gru_wi,              pk_wi, 3 * ND, ND, 13, 3);
    pack(gru_wh,              pk_wh, 3 * ND, ND, 13, 3);
    pack(lw,                  pk_lw, ND, 2 * ND, 5, 5);
    const int dims[4][2]  = {{3, 64}, {64, 32}, {32, 64}, {64, 3}};
    const int cipads[4]   = {32, 64, 32, 64};
    const int copads[4]   = {64, 32, 64, 32};
    for (int i = 0; i < 4; ++i) {
        int ci = dims[i][0], co = dims[i][1];
        int Nt = copads[i] / 16, KK = cipads[i] / 32;
        pack((const float*)d_in[4 + 8 * i], pk_cw[i], co, ci, Nt, KK);
        pack((const float*)d_in[8 + 8 * i], pk_rw[i], co, ci, Nt, KK);
    }

    // ---------------- encoder ----------------
    transpose_in<<<(MROWS * 32 + 255) / 256, 256, 0, stream>>>(x, buf0);
    float* cur = buf0;
    float* nxt = buf2;
    const int convBlocks = (MROWS / 16 + 7) / 8;   // one wave per 16-row M-tile
    for (int i = 0; i < 4; ++i) {
        int co = dims[i][1];
        int cip = cipads[i];
        gcn_spatial<<<BATCH, 256, 0, stream>>>(
            cur, buf1, Aall + (size_t)i * T_IN * NV * NV,
            Tmall + (size_t)i * NV * T_IN * T_IN, cip);
        const float* cb_ = (const float*)d_in[5 + 8 * i];
        const float* g_  = (const float*)d_in[6 + 8 * i];
        const float* bt_ = (const float*)d_in[7 + 8 * i];
        const float* rb_ = (const float*)d_in[9 + 8 * i];
        const float* rg_ = (const float*)d_in[10 + 8 * i];
        const float* rbt_= (const float*)d_in[11 + 8 * i];
        if (i == 0)
            gcn_conv_wmma<4, 1><<<convBlocks, 256, 0, stream>>>(
                buf1, cur, pk_cw[i], pk_rw[i], cb_, g_, bt_, rb_, rg_, rbt_,
                pre + i, nxt, co);
        else if (i == 1)
            gcn_conv_wmma<2, 2><<<convBlocks, 256, 0, stream>>>(
                buf1, cur, pk_cw[i], pk_rw[i], cb_, g_, bt_, rb_, rg_, rbt_,
                pre + i, nxt, co);
        else if (i == 2)
            gcn_conv_wmma<4, 1><<<convBlocks, 256, 0, stream>>>(
                buf1, cur, pk_cw[i], pk_rw[i], cb_, g_, bt_, rb_, rg_, rbt_,
                pre + i, nxt, co);
        else
            gcn_conv_wmma<2, 2><<<convBlocks, 256, 0, stream>>>(
                buf1, cur, pk_cw[i], pk_rw[i], cb_, g_, bt_, rb_, rg_, rbt_,
                pre + i, nxt, co);
        float* t = cur; cur = nxt; nxt = t;
    }
    make_enc<<<(BATCH * T_IN * 96 + 255) / 256, 256, 0, stream>>>(cur, enc);
    init_state<<<(BATCH * 96 + 255) / 256, 256, 0, stream>>>(enc, x, hs, tp);

    // ---------------- K, V projections ----------------
    gemm(enc, 96, pk_wk, qkv_b + ND,     nullptr, 0, kb, ND, 0, nullptr, 0,
         BATCH * T_IN, 5, 3, ND, 1.f);
    gemm(enc, 96, pk_wv, qkv_b + 2 * ND, nullptr, 0, vb, ND, 0, nullptr, 0,
         BATCH * T_IN, 5, 3, ND, 1.f);

    // ---------------- decoder scan: 25 serial steps ----------------
    const float scale = 0.21320071635561041f;  // 1/sqrt(22)
    for (int s = 0; s < T_OUT; ++s) {
        gemm(tp, 96, pk_wi, gru_bi, nullptr, 0, gi, 3 * ND, 0, nullptr, 0,
             BATCH, 13, 3, 3 * ND, 1.f);
        gemm(hs, 96, pk_wh, gru_bh, nullptr, 0, gh, 3 * ND, 0, nullptr, 0,
             BATCH, 13, 3, 3 * ND, 1.f);
        gru_gates<<<(BATCH * ND + 255) / 256, 256, 0, stream>>>(gi, gh, hs, cat);
        gemm(hs, 96, pk_wq, qkv_b, nullptr, 0, qb, ND, 0, nullptr, 0,
             BATCH, 5, 3, ND, scale);
        attn_kernel<<<(BATCH * NH + 255) / 256, 256, 0, stream>>>(qb, kb, vb, ctx);
        // ctx2 -> cat[:, 66:132] (store-zero mode keeps cols 132..145 zero)
        gemm(ctx, 96, pk_ow, obias, nullptr, 0, cat + ND, 160, 1, nullptr, 0,
             BATCH, 5, 3, ND, 1.f);
        // pred = cat @ lw.T + lb + tp -> out[:, s, :] and next tp (pad-zeroed)
        gemm(cat, 160, pk_lw, lb, tp, 96, out + (size_t)s * ND, T_OUT * ND, 0,
             tp, 96, BATCH, 5, 5, ND, 1.f);
    }
}
